// CaptionHead_62792421867532
// MI455X (gfx1250) — compile-verified, matrix operand order checked by reading, not run on hardware
//
#include <hip/hip_runtime.h>
#include <hip/hip_fp16.h>
#include <math.h>

typedef __attribute__((ext_vector_type(4)))  float    v4f;
typedef __attribute__((ext_vector_type(8)))  float    v8f;
typedef __attribute__((ext_vector_type(8)))  _Float16 v8h;
typedef __attribute__((ext_vector_type(16))) _Float16 v16h;

#define D_DIM 512          // feature dim (K)
#define C_DIM 1024         // captions (N)
#define KT    (D_DIM / 32) // 16 K-tiles of 32

// ---------------------------------------------------------------------------
// Kernel 1: voxel->point gather + L2 normalize + f32->f16.
// One wave32 per point row; each lane owns 16 contiguous floats (64B).
// ---------------------------------------------------------------------------
__global__ void gather_norm_f16(const float* __restrict__ feats,
                                const int* __restrict__ v2p,
                                _Float16* __restrict__ a16, int npts) {
    const int p    = (blockIdx.x * blockDim.x + threadIdx.x) >> 5;
    const int lane = threadIdx.x & 31;
    if (p >= npts) return;
    const float* src = feats + (size_t)v2p[p] * D_DIM + lane * 16;
    v4f f[4];
#pragma unroll
    for (int k = 0; k < 4; ++k) f[k] = *(const v4f*)(src + 4 * k);
    float ss = 0.f;
#pragma unroll
    for (int k = 0; k < 4; ++k)
        ss += f[k].x * f[k].x + f[k].y * f[k].y + f[k].z * f[k].z + f[k].w * f[k].w;
#pragma unroll
    for (int off = 16; off >= 1; off >>= 1) ss += __shfl_xor(ss, off, 32);
    const float inv = 1.0f / fmaxf(sqrtf(ss), 1e-12f);
    v8h h0, h1;
#pragma unroll
    for (int e = 0; e < 4; ++e) {
        h0[e]     = (_Float16)(f[0][e] * inv);
        h0[e + 4] = (_Float16)(f[1][e] * inv);
        h1[e]     = (_Float16)(f[2][e] * inv);
        h1[e + 4] = (_Float16)(f[3][e] * inv);
    }
    _Float16* dst = a16 + (size_t)p * D_DIM + lane * 16;
    *(v8h*)(dst)     = h0;
    *(v8h*)(dst + 8) = h1;
}

// ---------------------------------------------------------------------------
// Kernel 2: caption_embed f32 -> f16 (1 MB; stays resident in L2).
// ---------------------------------------------------------------------------
__global__ void cvt_cap_f16(const float* __restrict__ in,
                            _Float16* __restrict__ out, int n) {
    int i = blockIdx.x * blockDim.x + threadIdx.x;
    if (i < n) out[i] = (_Float16)in[i];
}

// ---------------------------------------------------------------------------
// WMMA fragment load: row-major f16 matrix, leading dim D_DIM.
// CDNA5 16-bit A/B layout (ISA 7.12.2): lane l holds row/col (l&15);
// lanes 0-15 carry K {k0..k0+7, k0+16..k0+23}, lanes 16-31 carry
// K {k0+8..k0+15, k0+24..k0+31}; two b128 loads per lane.
// ---------------------------------------------------------------------------
__device__ __forceinline__ v16h load_frag(const _Float16* __restrict__ base,
                                          int rc0, int k0, int lane) {
    const _Float16* p =
        base + (size_t)(rc0 + (lane & 15)) * D_DIM + (k0 + ((lane >> 4) << 3));
    v8h lo = *(const v8h*)(p);       // K = k0+8g .. +7
    v8h hi = *(const v8h*)(p + 16);  // K = k0+16+8g .. +7
    return __builtin_shufflevector(lo, hi, 0, 1, 2, 3, 4, 5, 6, 7,
                                   8, 9, 10, 11, 12, 13, 14, 15);
}

// ---------------------------------------------------------------------------
// Kernel 3: scores = (A_f16 @ B_f16^T) * exp(logit_scale), f32 out.
// 256 threads = 8 waves arranged 2(M) x 4(N); each wave owns 64x64 output
// (4x4 fragments of v_wmma_f32_16x16x32_f16), K-loop over 16 tiles.
// Out-of-range M tiles are CLAMPED for loads (wave-uniform, no EXEC churn
// in the hot loop) and only the stores are guarded.
// ---------------------------------------------------------------------------
__global__ void __launch_bounds__(256, 1)
wmma_scores(const _Float16* __restrict__ A, const _Float16* __restrict__ B,
            const float* __restrict__ lscale, float* __restrict__ S,
            int mtiles) {
    const int lane  = threadIdx.x & 31;
    const int wid   = threadIdx.x >> 5;
    const int waveM = wid >> 2;   // 0..1
    const int waveN = wid & 3;    // 0..3
    const int mf0 = blockIdx.y * 8 + waveM * 4;   // first M fragment (16 rows each)
    const int nf0 = blockIdx.x * 16 + waveN * 4;  // first N fragment

    int  mt[4];
    bool st[4];
#pragma unroll
    for (int i = 0; i < 4; ++i) {
        const int m = mf0 + i;
        st[i] = (m < mtiles);
        mt[i] = st[i] ? m : (mtiles - 1);  // clamp: loads always legal
    }

    v8f acc[4][4] = {};

    for (int kt = 0; kt < KT; ++kt) {
        const int k0 = kt * 32;
        if (kt + 1 < KT) {  // warm next K-tile of A (global_prefetch_b8)
            __builtin_prefetch(A + (size_t)(mt[0] * 16 + (lane & 15)) * D_DIM +
                                   (k0 + 32), 0, 1);
        }
        v16h b[4], a[4];
#pragma unroll
        for (int j = 0; j < 4; ++j) b[j] = load_frag(B, (nf0 + j) * 16, k0, lane);
#pragma unroll
        for (int i = 0; i < 4; ++i) a[i] = load_frag(A, mt[i] * 16, k0, lane);
#pragma unroll
        for (int i = 0; i < 4; ++i)
#pragma unroll
            for (int j = 0; j < 4; ++j)
                acc[i][j] = __builtin_amdgcn_wmma_f32_16x16x32_f16(
                    false, a[i], false, b[j], (short)0, acc[i][j], false, false);
    }

    const float sc = expf(lscale[0]);
    // C/D fragment layout: M = vgpr + 8*(lane>=16), N = lane&15
    const int nsub = lane & 15;
    const int msub = (lane >> 4) * 8;
#pragma unroll
    for (int i = 0; i < 4; ++i) {
        if (!st[i]) continue;
        const int row0 = mt[i] * 16 + msub;
#pragma unroll
        for (int j = 0; j < 4; ++j) {
            const int col = (nf0 + j) * 16 + nsub;
#pragma unroll
            for (int v = 0; v < 8; ++v)
                __builtin_nontemporal_store(acc[i][j][v] * sc,
                    S + (size_t)(row0 + v) * C_DIM + col);
        }
    }
}

// ---------------------------------------------------------------------------
// Kernel 4: per-row softmax offset off[row] = max + log(sum exp(x - max)).
// One block per row; values live in registers, shuffle+LDS reductions.
// Writing only 1 float/row avoids a full 614MB logp materialization.
// ---------------------------------------------------------------------------
__global__ void row_logsumexp(const float* __restrict__ S,
                              float* __restrict__ off_out, int rows) {
    const int row = blockIdx.x;
    if (row >= rows) return;
    const float* x = S + (size_t)row * C_DIM;
    const int tid  = threadIdx.x;
    const int lane = tid & 31;
    const int wid  = tid >> 5;
    __shared__ float redm[8];
    __shared__ float reds[8];

    float v[4];
    float m = -INFINITY;
#pragma unroll
    for (int k = 0; k < 4; ++k) {
        v[k] = __builtin_nontemporal_load(x + tid + 256 * k);
        m = fmaxf(m, v[k]);
    }
#pragma unroll
    for (int off = 16; off >= 1; off >>= 1) m = fmaxf(m, __shfl_xor(m, off, 32));
    if (lane == 0) redm[wid] = m;
    __syncthreads();
    float bm = redm[0];
#pragma unroll
    for (int w = 1; w < 8; ++w) bm = fmaxf(bm, redm[w]);

    float s = 0.f;
#pragma unroll
    for (int k = 0; k < 4; ++k) s += __expf(v[k] - bm);
#pragma unroll
    for (int off = 16; off >= 1; off >>= 1) s += __shfl_xor(s, off, 32);
    if (lane == 0) reds[wid] = s;
    __syncthreads();
    float bs = 0.f;
#pragma unroll
    for (int w = 0; w < 8; ++w) bs += reds[w];
    if (tid == 0) off_out[row] = bm + logf(bs);
}

// ---------------------------------------------------------------------------
// Kernel 5: segment-mean pooling on raw scores with the log-softmax folded in:
//   pooled[s,c] = ( sum_t scores[idx[t],c] - sum_t off[idx[t]] ) / cnt
// One block per caption; binary search the sorted segment_ids for [lo,hi),
// stage indices+offsets in LDS, 4 columns/thread.
// ---------------------------------------------------------------------------
__device__ __forceinline__ int lower_bound_i(const int* __restrict__ a, int n, int key) {
    int lo = 0, hi = n;
    while (lo < hi) {
        int mid = (lo + hi) >> 1;
        if (a[mid] < key) lo = mid + 1; else hi = mid;
    }
    return lo;
}

__global__ void pool_by_segment(const float* __restrict__ scores,
                                const float* __restrict__ off,
                                const int* __restrict__ c2p,
                                const int* __restrict__ seg, int T,
                                float* __restrict__ pooled,
                                float* __restrict__ counts) {
    const int s   = blockIdx.x;
    const int tid = threadIdx.x;
    const int lo = lower_bound_i(seg, T, s);
    const int hi = lower_bound_i(seg, T, s + 1);
    __shared__ int   sidx[256];
    __shared__ float soff[256];
    float acc[4] = {0.f, 0.f, 0.f, 0.f};
    float offacc = 0.f;
    for (int base = lo; base < hi; base += 256) {
        const int n = min(256, hi - base);
        if (tid < n) {
            const int p = c2p[base + tid];
            sidx[tid] = p;
            soff[tid] = off[p];
        }
        __syncthreads();
        for (int j = 0; j < n; ++j) {
            const float* r = scores + (size_t)sidx[j] * C_DIM;
#pragma unroll
            for (int k = 0; k < 4; ++k) acc[k] += r[tid + 256 * k];
            offacc += soff[j];
        }
        __syncthreads();
    }
    const float cnt = (float)(hi - lo);
    const float inv = 1.0f / fmaxf(cnt, 1.0f);
#pragma unroll
    for (int k = 0; k < 4; ++k)
        pooled[(size_t)s * C_DIM + tid + 256 * k] = (acc[k] - offacc) * inv;
    if (tid == 0) counts[s] = cnt;
}

// ---------------------------------------------------------------------------
// Host launcher
// ---------------------------------------------------------------------------
extern "C" void kernel_launch(void* const* d_in, const int* in_sizes, int n_in,
                              void* d_out, int out_size, void* d_ws, size_t ws_size,
                              hipStream_t stream) {
    const float* adapter = (const float*)d_in[0];  // [N_vox, 512]
    const float* cap     = (const float*)d_in[1];  // [1024, 512], unit rows
    const float* lscale  = (const float*)d_in[2];  // [1]
    const int*   v2p     = (const int*)d_in[3];    // [N_pts]
    const int*   c2p     = (const int*)d_in[4];    // [T]
    const int*   seg     = (const int*)d_in[5];    // [T] sorted
    // d_in[6] = num_captions (device scalar); shapes fixed at C_DIM/D_DIM.

    const int npts   = in_sizes[3];
    const int T      = in_sizes[4];
    const int mtiles = (npts + 15) / 16;

    float* pooled = (float*)d_out;                      // [1024,1024]
    float* counts = pooled + (size_t)C_DIM * C_DIM;     // [1024]

    // Workspace carve-up (256B aligned):
    char* ws = (char*)d_ws;
    size_t a_bytes = ((size_t)mtiles * 16 * D_DIM * sizeof(_Float16) + 255) & ~(size_t)255;
    size_t b_bytes = ((size_t)C_DIM * D_DIM * sizeof(_Float16) + 255) & ~(size_t)255;
    size_t s_bytes = ((size_t)mtiles * 16 * C_DIM * sizeof(float) + 255) & ~(size_t)255;
    _Float16* a16    = (_Float16*)ws;
    _Float16* b16    = (_Float16*)(ws + a_bytes);
    float*    scores = (float*)(ws + a_bytes + b_bytes);            // [npts, 1024]
    float*    offrow = (float*)(ws + a_bytes + b_bytes + s_bytes);  // [npts]

    // 1) gather + normalize + fp16 (one wave per point row)
    gather_norm_f16<<<(npts + 7) / 8, 256, 0, stream>>>(adapter, v2p, a16, npts);
    // 2) caption matrix to fp16
    cvt_cap_f16<<<(C_DIM * D_DIM + 255) / 256, 256, 0, stream>>>(cap, b16, C_DIM * D_DIM);
    // 3) WMMA GEMM + logit scale
    dim3 grid(C_DIM / 256, (mtiles + 7) / 8);
    wmma_scores<<<grid, 256, 0, stream>>>(a16, b16, lscale, scores, mtiles);
    // 4) per-row logsumexp offsets (1 float/row instead of 4KB/row)
    row_logsumexp<<<npts, 256, 0, stream>>>(scores, offrow, npts);
    // 5) segment-mean pooling with softmax offset folded in
    pool_by_segment<<<C_DIM, 256, 0, stream>>>(scores, offrow, c2p, seg, T,
                                               pooled, counts);
}
